// MiddleMENoDownsampleXY_39685497815877
// MI455X (gfx1250) — matrix operand order repaired and checked by reading, not compiled, not run on hardware
//
#include <hip/hip_runtime.h>
#include <hip/hip_bf16.h>

typedef __attribute__((ext_vector_type(16))) _Float16 v16h;
typedef __attribute__((ext_vector_type(8)))  _Float16 v8h;
typedef __attribute__((ext_vector_type(8)))  float    v8f;

#define HH 128
#define WW 128
#define HP 130   // padded H (1-voxel zero halo)
#define WP 130   // padded W

// ---------------------------------------------------------------------------
// Scatter sparse voxels into dense padded f16 grid [z+1][y+1][x+1][c16] + occ
// ---------------------------------------------------------------------------
__global__ void scatter_kernel(const float* __restrict__ vf,
                               const int* __restrict__ coors,
                               _Float16* __restrict__ grid0,
                               float* __restrict__ occ16, int N) {
    int tid = blockIdx.x * blockDim.x + threadIdx.x;
    if (tid >= N * 16) return;
    int i = tid >> 4, c = tid & 15;
    int z = coors[i * 4 + 1], y = coors[i * 4 + 2], x = coors[i * 4 + 3];
    int ppos = ((z + 1) * HP + (y + 1)) * WP + (x + 1);
    grid0[ppos * 16 + c] = (_Float16)vf[i * 16 + c];
    if (c == 0) occ16[(z * HH + y) * WW + x] = 1.0f;
}

// occ z-pool: out[z] = max(in[2z], in[2z+1])   (unpadded occ arrays)
__global__ void pool_kernel(const float* __restrict__ in, float* __restrict__ out, int total) {
    int tid = blockIdx.x * blockDim.x + threadIdx.x;
    if (tid >= total) return;
    int x = tid % WW, y = (tid / WW) % HH, z = tid / (WW * HH);
    out[tid] = fmaxf(in[((2 * z) * HH + y) * WW + x], in[((2 * z + 1) * HH + y) * WW + x]);
}

// ---------------------------------------------------------------------------
// Fold BN into per-channel scale/bias: sb[layer*128 + c] = scale, [+64] = bias
// layers: 0,1 -> bn32 rows 0,1 ; 2..10 -> bn64 rows 0..8
// ---------------------------------------------------------------------------
__global__ void bn_kernel(const float* g32, const float* b32, const float* m32, const float* v32,
                          const float* g64, const float* b64, const float* m64, const float* v64,
                          float* __restrict__ sb) {
    int tid = blockIdx.x * blockDim.x + threadIdx.x;
    if (tid >= 11 * 64) return;
    int layer = tid / 64, c = tid % 64;
    float g = 1.f, b = 0.f, m = 0.f, v = 1.f;
    if (layer < 2) {
        if (c < 32) { g = g32[layer*32+c]; b = b32[layer*32+c]; m = m32[layer*32+c]; v = v32[layer*32+c]; }
    } else {
        int r = layer - 2;
        g = g64[r*64+c]; b = b64[r*64+c]; m = m64[r*64+c]; v = v64[r*64+c];
    }
    float inv = g * rsqrtf(v + 1e-5f);
    sb[layer * 128 + c] = inv;
    sb[layer * 128 + 64 + c] = b - m * inv;
}

// ---------------------------------------------------------------------------
// Pre-swizzle conv weights (OIDHW f32) into WMMA A-fragment layout, f16.
// frag linear index = (((cot*T + t)*CIB + b)*32 + lane)*16 + i
// A(m,k): lane = m + 16*(k/16), i = k%16
// ---------------------------------------------------------------------------
__global__ void wprep_kernel(const float* __restrict__ w, _Float16* __restrict__ frag,
                             int Cin, int T, int total) {
    int tid = blockIdx.x * blockDim.x + threadIdx.x;
    if (tid >= total) return;
    int i    = tid & 15;
    int lane = (tid >> 4) & 31;
    int rest = tid >> 9;
    int CIB  = (Cin + 31) >> 5;
    int b    = rest % CIB;  rest /= CIB;
    int t    = rest % T;
    int cot  = rest / T;
    int m  = lane & 15, kg = lane >> 4;
    int k  = (kg << 4) + i;
    int co = cot * 16 + m;
    int ci = b * 32 + k;
    float val = (ci < Cin) ? w[(co * Cin + ci) * T + t] : 0.0f;
    frag[tid] = (_Float16)val;
}

// ---------------------------------------------------------------------------
// Implicit-GEMM conv3d + BN + ReLU + occ via v_wmma_f32_16x16x32_f16.
// Activations live in zero-halo padded buffers [(D+2)][130][130][C] so the
// inner loop has NO boundary clamps/selects: address -> load -> WMMA.
// One wave: OUTC channels x 32 x-positions (2 N-tiles) at one (zo, y) site.
// B(k,n): lane = n + 16*(k/16), half = k%16  -> contiguous 32B per lane
// D(m,n): vgpr r, lane group g -> m = r + 8g
// ---------------------------------------------------------------------------
template<int INC, int OUTC, int KD, int KHW, int SZ, bool FINAL>
__global__ __launch_bounds__(256) void conv_wmma(
    const _Float16* __restrict__ in, _Float16* __restrict__ out, float* __restrict__ fout,
    const _Float16* __restrict__ wfrag, const float* __restrict__ sb,
    const float* __restrict__ occ, int outD)
{
    constexpr int CIB = (INC + 31) / 32;
    constexpr int COT = OUTC / 16;
    constexpr int T   = KD * KHW * KHW;

    int wave = blockIdx.x * (blockDim.x >> 5) + (threadIdx.x >> 5);
    int lane = threadIdx.x & 31;
    int n = lane & 15, g = lane >> 4;
    int xp = wave & 3;            // 4 double-tiles of 32 along x
    int y  = (wave >> 2) & 127;
    int zo = wave >> 9;
    if (zo >= outD) return;
    int x0 = xp << 5;

    v8f zf = {};
    v8f acc[2][COT];
#pragma unroll
    for (int u = 0; u < 2; ++u)
#pragma unroll
        for (int c = 0; c < COT; ++c) acc[u][c] = zf;

    // per-lane padded x coordinate of tap kw: px = x + kw - KHW/2 + 1
    int pxb = x0 + n + 1 - (KHW >> 1);

    for (int kd = 0; kd < KD; ++kd) {
        int pz = zo * SZ + kd;          // padded z (KD==3: -1+1 == 0 offset)
        for (int kh = 0; kh < KHW; ++kh) {
            int py = y + kh + 1 - (KHW >> 1);
            const _Float16* row = in + ((pz * HP + py) * WP) * INC + g * 16;
#pragma unroll
            for (int kw = 0; kw < KHW; ++kw) {
                int t = (kd * KHW + kh) * KHW + kw;
                const _Float16* bp0 = row + (pxb + kw) * INC;
                const _Float16* bp1 = bp0 + 16 * INC;
#pragma unroll
                for (int b = 0; b < CIB; ++b) {
                    v16h bv0 = *(const v16h*)(bp0 + b * 32);
                    v16h bv1 = *(const v16h*)(bp1 + b * 32);
#pragma unroll
                    for (int c = 0; c < COT; ++c) {
                        v16h av = *(const v16h*)(wfrag + (((c * T + t) * CIB + b) * 32 + lane) * 16);
                        acc[0][c] = __builtin_amdgcn_wmma_f32_16x16x32_f16(
                            false, av, false, bv0, (short)0, acc[0][c], false, false);
                        acc[1][c] = __builtin_amdgcn_wmma_f32_16x16x32_f16(
                            false, av, false, bv1, (short)0, acc[1][c], false, false);
                    }
                }
            }
        }
    }

#pragma unroll
    for (int u = 0; u < 2; ++u) {
        int x = x0 + u * 16 + n;
        float oc = occ[(zo * HH + y) * WW + x];
        if (!FINAL) {
            size_t ppos = (size_t)(((zo + 1) * HP + (y + 1)) * WP + (x + 1));
            _Float16* op = out + ppos * OUTC + g * 8;
#pragma unroll
            for (int c = 0; c < COT; ++c) {
                v8h hv;
#pragma unroll
                for (int r = 0; r < 8; ++r) {
                    int co = c * 16 + g * 8 + r;
                    float v = acc[u][c][r] * sb[co] + sb[64 + co];
                    v = v > 0.f ? v : 0.f;
                    hv[r] = (_Float16)(v * oc);
                }
                *(v8h*)(op + c * 16) = hv;
            }
        } else {
#pragma unroll
            for (int c = 0; c < COT; ++c) {
#pragma unroll
                for (int r = 0; r < 8; ++r) {
                    int co = c * 16 + g * 8 + r;
                    float v = acc[u][c][r] * sb[co] + sb[64 + co];
                    v = v > 0.f ? v : 0.f;
                    fout[((co * outD + zo) * HH + y) * WW + x] = v * oc;
                }
            }
        }
    }
}

// ---------------------------------------------------------------------------
extern "C" void kernel_launch(void* const* d_in, const int* in_sizes, int n_in,
                              void* d_out, int out_size, void* d_ws, size_t ws_size,
                              hipStream_t stream) {
    const float* vf   = (const float*)d_in[0];
    const int*   co   = (const int*)d_in[1];
    const float* w0   = (const float*)d_in[2];
    const float* w1   = (const float*)d_in[3];
    const float* w2   = (const float*)d_in[4];
    const float* w64p = (const float*)d_in[5];
    const float* w10  = (const float*)d_in[6];
    const float* g32  = (const float*)d_in[7];
    const float* b32  = (const float*)d_in[8];
    const float* m32  = (const float*)d_in[9];
    const float* v32  = (const float*)d_in[10];
    const float* g64  = (const float*)d_in[11];
    const float* b64  = (const float*)d_in[12];
    const float* m64  = (const float*)d_in[13];
    const float* v64  = (const float*)d_in[14];
    int N = in_sizes[0] / 16;

    // ---- workspace layout ----
    char* ws = (char*)d_ws;
    size_t off = 0;
    auto alloc = [&](size_t bytes) -> char* {
        char* p = ws + off;
        off += (bytes + 255) & ~(size_t)255;
        return p;
    };
    // padded buffer sizes (bytes), +64B tail pad for INC=16 g=1 over-read
    size_t grid0_bytes = (size_t)18 * HP * WP * 16 * 2 + 64;                 // 16ch, D16 padded
    size_t buf_bytes   = (size_t)10 * HP * WP * 64 * 2 + 64;                 // 64ch D10 padded
    size_t buf32_bytes = (size_t)18 * HP * WP * 32 * 2 + 64;                 // 32ch D18 padded
    size_t bufmax      = buf_bytes > buf32_bytes ? buf_bytes : buf32_bytes;
    _Float16* grid0 = (_Float16*)alloc(grid0_bytes);
    _Float16* bufA  = (_Float16*)alloc(bufmax);
    _Float16* bufB  = (_Float16*)alloc(bufmax);
    float* occ16 = (float*)alloc((size_t)16 * HH * WW * 4);
    float* occ8  = (float*)alloc((size_t)8  * HH * WW * 4);
    float* occ4  = (float*)alloc((size_t)4  * HH * WW * 4);
    float* occ2  = (float*)alloc((size_t)2  * HH * WW * 4);
    float* sb    = (float*)alloc((size_t)11 * 128 * 4);

    // weight fragment halves per layer: COT*T*CIB*512
    const int fh0  = 2 * 27 * 1 * 512;   // L0  16->32
    const int fh1  = 2 * 27 * 1 * 512;   // L1  32->32
    const int fh2  = 4 * 27 * 1 * 512;   // L2  32->64
    const int fh64 = 4 * 27 * 2 * 512;   // L3..L9 64->64
    const int fh10 = 4 * 3  * 2 * 512;   // L10 64->64 (3,1,1)
    _Float16* frag[11];
    frag[0] = (_Float16*)alloc((size_t)fh0 * 2);
    frag[1] = (_Float16*)alloc((size_t)fh1 * 2);
    frag[2] = (_Float16*)alloc((size_t)fh2 * 2);
    for (int i = 0; i < 7; ++i) frag[3 + i] = (_Float16*)alloc((size_t)fh64 * 2);
    frag[10] = (_Float16*)alloc((size_t)fh10 * 2);

    // ---- prep: zero padded input grid + occ; zero ping-pong for first layouts
    hipMemsetAsync(grid0, 0, grid0_bytes, stream);
    hipMemsetAsync(occ16, 0, (size_t)16 * HH * WW * 4, stream);
    hipMemsetAsync(bufA, 0, bufmax, stream);
    hipMemsetAsync(bufB, 0, bufmax, stream);

    scatter_kernel<<<(N * 16 + 255) / 256, 256, 0, stream>>>(vf, co, grid0, occ16, N);
    pool_kernel<<<(8 * HH * WW + 255) / 256, 256, 0, stream>>>(occ16, occ8, 8 * HH * WW);
    pool_kernel<<<(4 * HH * WW + 255) / 256, 256, 0, stream>>>(occ8, occ4, 4 * HH * WW);
    pool_kernel<<<(2 * HH * WW + 255) / 256, 256, 0, stream>>>(occ4, occ2, 2 * HH * WW);
    bn_kernel<<<(11 * 64 + 255) / 256, 256, 0, stream>>>(g32, b32, m32, v32, g64, b64, m64, v64, sb);

    wprep_kernel<<<(fh0 + 255) / 256, 256, 0, stream>>>(w0, frag[0], 16, 27, fh0);
    wprep_kernel<<<(fh1 + 255) / 256, 256, 0, stream>>>(w1, frag[1], 32, 27, fh1);
    wprep_kernel<<<(fh2 + 255) / 256, 256, 0, stream>>>(w2, frag[2], 32, 27, fh2);
    for (int i = 0; i < 7; ++i)
        wprep_kernel<<<(fh64 + 255) / 256, 256, 0, stream>>>(w64p + (size_t)i * 64 * 64 * 27,
                                                             frag[3 + i], 64, 27, fh64);
    wprep_kernel<<<(fh10 + 255) / 256, 256, 0, stream>>>(w10, frag[10], 64, 3, fh10);

    // ---- conv pipeline (blocks = outD*64, 8 waves/block, wave = 32-wide x strip)
    // Re-zero a ping-pong buffer whenever its layout (C or D) changes, so the
    // zero-halo invariant holds; stream order puts each memset after the last
    // reader of the old contents.
    conv_wmma<16, 32, 3, 3, 1, false><<<16 * 64, 256, 0, stream>>>(grid0, bufA, nullptr, frag[0], sb + 0 * 128, occ16, 16);
    conv_wmma<32, 32, 3, 3, 1, false><<<16 * 64, 256, 0, stream>>>(bufA, bufB, nullptr, frag[1], sb + 1 * 128, occ16, 16);
    hipMemsetAsync(bufA, 0, bufmax, stream);  // 32ch/D18 -> 64ch/D10
    conv_wmma<32, 64, 3, 3, 2, false><<< 8 * 64, 256, 0, stream>>>(bufB, bufA, nullptr, frag[2], sb + 2 * 128, occ8, 8);
    hipMemsetAsync(bufB, 0, bufmax, stream);  // 32ch/D18 -> 64ch/D10
    conv_wmma<64, 64, 3, 3, 1, false><<< 8 * 64, 256, 0, stream>>>(bufA, bufB, nullptr, frag[3], sb + 3 * 128, occ8, 8);
    conv_wmma<64, 64, 3, 3, 1, false><<< 8 * 64, 256, 0, stream>>>(bufB, bufA, nullptr, frag[4], sb + 4 * 128, occ8, 8);
    conv_wmma<64, 64, 3, 3, 1, false><<< 8 * 64, 256, 0, stream>>>(bufA, bufB, nullptr, frag[5], sb + 5 * 128, occ8, 8);
    hipMemsetAsync(bufA, 0, bufmax, stream);  // 64ch/D10 -> 64ch/D6
    conv_wmma<64, 64, 3, 3, 2, false><<< 4 * 64, 256, 0, stream>>>(bufB, bufA, nullptr, frag[6], sb + 6 * 128, occ4, 4);
    hipMemsetAsync(bufB, 0, bufmax, stream);  // 64ch/D10 -> 64ch/D6
    conv_wmma<64, 64, 3, 3, 1, false><<< 4 * 64, 256, 0, stream>>>(bufA, bufB, nullptr, frag[7], sb + 7 * 128, occ4, 4);
    conv_wmma<64, 64, 3, 3, 1, false><<< 4 * 64, 256, 0, stream>>>(bufB, bufA, nullptr, frag[8], sb + 8 * 128, occ4, 4);
    conv_wmma<64, 64, 3, 3, 1, false><<< 4 * 64, 256, 0, stream>>>(bufA, bufB, nullptr, frag[9], sb + 9 * 128, occ4, 4);
    conv_wmma<64, 64, 3, 1, 2, true ><<< 2 * 64, 256, 0, stream>>>(bufB, nullptr, (float*)d_out, frag[10], sb + 10 * 128, occ2, 2);
}
